// DyGMMBlock_19310172963223
// MI455X (gfx1250) — compile-verified, matrix operand order checked by reading, not run on hardware
//
// DyGMMBlock forward for MI455X (gfx1250, wave32, WMMA).
// All heavy matmuls (QKV/FFN/scores/ctx/cross/final) run through one batched
// bf16 WMMA GEMM (v_wmma_f32_16x16x32_bf16, f32 accumulate). Weights are
// transposed+cast to bf16 once per call (bf16 weight set ~150MB fits in the
// 192MB L2). Softmax / LayerNorm / Gaussian masks are small VALU kernels.
// GEMM wave tile: 32Mx64N, K unrolled by 64 with double-buffered fragments
// (software pipelining), straight-line WMMA (no divergent guards -> EXEC
// stays all-ones and no exec-mask mov churn).
#include <hip/hip_runtime.h>
#include <hip/hip_bf16.h>

#define H_    768
#define NH_   12
#define DH_   64
#define B_    4
#define S_    512
#define FF_   3072
#define NBR_  11
#define MAP_  512

typedef __bf16 bf16;
typedef __attribute__((ext_vector_type(16))) __bf16 bf16x16;
typedef __attribute__((ext_vector_type(8)))  __bf16 bf16x8;
typedef __attribute__((ext_vector_type(8)))  float  f32x8;

__device__ __constant__ float c_wids[10] = {0.5f,1.0f,5.0f,10.0f,3.0f,0.1f,8.0f,0.05f,2.0f,15.0f};

__device__ inline bf16 f2bf(float f) {
  unsigned u = __builtin_bit_cast(unsigned, f);
  u += 0x7FFFu + ((u >> 16) & 1u);            // round-to-nearest-even
  unsigned short h = (unsigned short)(u >> 16);
  return __builtin_bit_cast(bf16, h);
}

__device__ inline float blk_reduce(float v, float* red, bool do_max) {
  int t = threadIdx.x;
  red[t] = v; __syncthreads();
  for (int s = 128; s > 0; s >>= 1) {
    if (t < s) red[t] = do_max ? fmaxf(red[t], red[t + s]) : (red[t] + red[t + s]);
    __syncthreads();
  }
  float r = red[0];
  __syncthreads();
  return r;
}

__device__ inline bf16x16 ldfrag(const bf16* p) {
  bf16x8 lo = *(const bf16x8*)p;
  bf16x8 hi = *(const bf16x8*)(p + 16);
  return __builtin_shufflevector(lo, hi, 0,1,2,3,4,5,6,7,8,9,10,11,12,13,14,15);
}

// ---------------------------------------------------------------------------
// Generic batched GEMM:  C[z][M,N] = A[z][M,K] * Bt[z][N,K]^T  (+bias)(+relu)
// A, Bt bf16; C f32 or bf16.  z1=z/subz, z2=z%subz, off = z1*s1 + z2*s2.
// Wave tile 32Mx64N: 8x v_wmma_f32_16x16x32_bf16 per K=32 step, K unrolled
// by 64 with two fragment buffers so loads overlap WMMA.
// Requires: K % 64 == 0, N % 64 == 0, lda/ldb % 8 == 0, 16B-aligned rows.
// ---------------------------------------------------------------------------
__global__ __launch_bounds__(256) void k_gemm(
    const bf16* __restrict__ A, long lda, long sA1, long sA2,
    const bf16* __restrict__ Bt, long ldb, long sB1, long sB2,
    void* __restrict__ Cv, long ldc, long sC1, long sC2, int c_is_bf16,
    const float* __restrict__ bias, int relu,
    int M, int N, int K, int subz)
{
  int z  = blockIdx.z;
  int z1 = z / subz, z2 = z % subz;
  const bf16* Ab = A  + (long)z1 * sA1 + (long)z2 * sA2;
  const bf16* Bb = Bt + (long)z1 * sB1 + (long)z2 * sB2;
  long cOff = (long)z1 * sC1 + (long)z2 * sC2;

  int wave = threadIdx.x >> 5;
  int lane = threadIdx.x & 31;
  int tilesN = N >> 6;                // N % 64 == 0
  int tilesM = (M + 31) >> 5;
  int tile = blockIdx.x * 8 + wave;
  if (tile >= tilesM * tilesN) return;
  int tm = tile / tilesN, tn = tile % tilesN;
  int m0 = tm << 5, n0 = tn << 6;

  int row = lane & 15;            // A row / B column / C column within tile
  int kg  = (lane >> 4) << 3;     // K sub-group base (0 or 8)

  int ar0 = m0 + row;      if (ar0 > M - 1) ar0 = M - 1;   // clamped; OOB rows never stored
  int ar1 = m0 + 16 + row; if (ar1 > M - 1) ar1 = M - 1;
  const bf16* Ap0 = Ab + (long)ar0 * lda + kg;
  const bf16* Ap1 = Ab + (long)ar1 * lda + kg;
  const bf16* Bp0 = Bb + (long)(n0 +  0 + row) * ldb + kg;
  const bf16* Bp1 = Bb + (long)(n0 + 16 + row) * ldb + kg;
  const bf16* Bp2 = Bb + (long)(n0 + 32 + row) * ldb + kg;
  const bf16* Bp3 = Bb + (long)(n0 + 48 + row) * ldb + kg;

  f32x8 acc[2][4];
#pragma unroll
  for (int u = 0; u < 2; ++u)
#pragma unroll
    for (int t = 0; t < 4; ++t)
      acc[u][t] = (f32x8){0.f,0.f,0.f,0.f,0.f,0.f,0.f,0.f};

  bf16x16 fa0[2], fa1[2], fb0[2], fb1[2], fb2[2], fb3[2];

#define LDFRAGS(buf, kk)                  \
  do {                                    \
    fa0[buf] = ldfrag(Ap0 + (kk));        \
    fa1[buf] = ldfrag(Ap1 + (kk));        \
    fb0[buf] = ldfrag(Bp0 + (kk));        \
    fb1[buf] = ldfrag(Bp1 + (kk));        \
    fb2[buf] = ldfrag(Bp2 + (kk));        \
    fb3[buf] = ldfrag(Bp3 + (kk));        \
  } while (0)

#define DOWMMA(buf)                                                           \
  do {                                                                        \
    acc[0][0] = __builtin_amdgcn_wmma_f32_16x16x32_bf16(                      \
        false, fa0[buf], false, fb0[buf], (short)0, acc[0][0], false, false); \
    acc[1][0] = __builtin_amdgcn_wmma_f32_16x16x32_bf16(                      \
        false, fa1[buf], false, fb0[buf], (short)0, acc[1][0], false, false); \
    acc[0][1] = __builtin_amdgcn_wmma_f32_16x16x32_bf16(                      \
        false, fa0[buf], false, fb1[buf], (short)0, acc[0][1], false, false); \
    acc[1][1] = __builtin_amdgcn_wmma_f32_16x16x32_bf16(                      \
        false, fa1[buf], false, fb1[buf], (short)0, acc[1][1], false, false); \
    acc[0][2] = __builtin_amdgcn_wmma_f32_16x16x32_bf16(                      \
        false, fa0[buf], false, fb2[buf], (short)0, acc[0][2], false, false); \
    acc[1][2] = __builtin_amdgcn_wmma_f32_16x16x32_bf16(                      \
        false, fa1[buf], false, fb2[buf], (short)0, acc[1][2], false, false); \
    acc[0][3] = __builtin_amdgcn_wmma_f32_16x16x32_bf16(                      \
        false, fa0[buf], false, fb3[buf], (short)0, acc[0][3], false, false); \
    acc[1][3] = __builtin_amdgcn_wmma_f32_16x16x32_bf16(                      \
        false, fa1[buf], false, fb3[buf], (short)0, acc[1][3], false, false); \
  } while (0)

  LDFRAGS(0, 0);
  for (int k0 = 0; k0 < K; k0 += 64) {
    LDFRAGS(1, k0 + 32);
    DOWMMA(0);
    if (k0 + 64 < K) LDFRAGS(0, k0 + 64);   // uniform (K is a kernel arg)
    DOWMMA(1);
  }
#undef LDFRAGS
#undef DOWMMA

  int mb = m0 + ((lane >> 4) << 3);  // C row base for this half-wave
#pragma unroll
  for (int t = 0; t < 4; ++t) {
    int n = n0 + t * 16 + row;
    float bv = bias ? bias[n] : 0.0f;
#pragma unroll
    for (int u = 0; u < 2; ++u) {
#pragma unroll
      for (int r = 0; r < 8; ++r) {
        int m = mb + u * 16 + r;
        if (m < M) {
          float c = acc[u][t][r] + bv;
          if (relu) c = fmaxf(c, 0.0f);
          long idx = cOff + (long)m * ldc + n;
          if (c_is_bf16) ((bf16*)Cv)[idx] = f2bf(c);
          else           ((float*)Cv)[idx] = c;
        }
      }
    }
  }
}

// fp32 (K,N) -> bf16 (N,K), LDS-tiled, batched over z.
__global__ __launch_bounds__(256) void k_transpose_w(
    const float* __restrict__ src, bf16* __restrict__ dst,
    int K, int N, long sstride, long dstride)
{
  __shared__ float t[32][33];
  int z = blockIdx.z;
  const float* s = src + (long)z * sstride;
  bf16* d = dst + (long)z * dstride;
  int kb = blockIdx.x * 32, nb = blockIdx.y * 32;
  int tx = threadIdx.x & 31, ty = threadIdx.x >> 5;
  for (int i = ty; i < 32; i += 8) {
    int k = kb + i, n = nb + tx;
    t[i][tx] = (k < K && n < N) ? s[(long)k * N + n] : 0.0f;
  }
  __syncthreads();
  for (int i = ty; i < 32; i += 8) {
    int n = nb + i, k = kb + tx;
    if (n < N && k < K) d[(long)n * K + k] = f2bf(t[tx][i]);
  }
}

// bf16 V (B,S,H) -> bf16 V^T (B*NH, DH, S)   grid (S/32, DH/32, B*NH)
__global__ __launch_bounds__(256) void k_transpose_v(
    const bf16* __restrict__ v, bf16* __restrict__ vT)
{
  __shared__ bf16 t[32][33];
  int z = blockIdx.z;
  int b = z / NH_, h = z % NH_;
  int s0 = blockIdx.x * 32, d0 = blockIdx.y * 32;
  int tx = threadIdx.x & 31, ty = threadIdx.x >> 5;
  const bf16* src = v + (long)b * S_ * H_ + h * DH_;
  for (int i = ty; i < 32; i += 8)
    t[i][tx] = src[(long)(s0 + i) * H_ + (d0 + tx)];
  __syncthreads();
  bf16* dst = vT + (long)z * DH_ * S_;
  for (int i = ty; i < 32; i += 8)
    dst[(long)(d0 + i) * S_ + (s0 + tx)] = t[tx][i];
}

__global__ void k_cvt(const float* __restrict__ s, bf16* __restrict__ d, long n) {
  long i = (long)blockIdx.x * 256 + threadIdx.x;
  if (i < n) d[i] = f2bf(s[i]);
}

__global__ void k_bcast_wt(const float* __restrict__ wt, bf16* __restrict__ wtb) {
  int i = blockIdx.x * 256 + threadIdx.x;
  if (i < B_ * H_) wtb[i] = f2bf(wt[i % H_]);
}

__global__ void k_maskadd(const float* __restrict__ am, float* __restrict__ ma, int n) {
  int i = blockIdx.x * 256 + threadIdx.x;
  if (i < n) ma[i] = (1.0f - am[i]) * -10000.0f;
}

// Gaussian masks, row-max-normalized.  grid (S, NBR), 256 threads.
__global__ __launch_bounds__(256) void k_gmm(float* __restrict__ g) {
  int i = blockIdx.x;
  int br = blockIdx.y;
  float* grow = g + ((long)br * S_ + i) * S_;
  if (br == 0) {
    for (int j = threadIdx.x; j < S_; j += 256) grow[j] = 1.0f;
    return;
  }
  float wid   = fmaxf(c_wids[br - 1], 0.01f);
  float width = wid / 9.0f;
  float center = (float)i / (float)S_;
  float inv2w2 = 1.0f / (2.0f * width * width);
  float e[2]; float mx = -1e30f;
#pragma unroll
  for (int t = 0; t < 2; ++t) {
    int j = threadIdx.x + t * 256;
    float pos = (float)j / (float)(S_ - 1);
    float d = pos - center;
    e[t] = -d * d * inv2w2;
    mx = fmaxf(mx, e[t]);
  }
  __shared__ float red[256];
  mx = blk_reduce(mx, red, true);
#pragma unroll
  for (int t = 0; t < 2; ++t)
    grow[threadIdx.x + t * 256] = __expf(e[t] - mx);
}

// Fused: v = scores*scale * gmm + mask_add ; softmax over 512 ; bf16 probs.
// grid = #rows, 256 threads.  g==nullptr for cross-attention (no mask mult).
__global__ __launch_bounds__(256) void k_softmax(
    const float* __restrict__ scores, const float* __restrict__ g, int q_rows,
    const float* __restrict__ maskadd, int rows_per_batch,
    float scale, bf16* __restrict__ probs)
{
  long r = blockIdx.x;
  const float* srow = scores + r * S_;
  const float* grow = g ? g + (long)(r % q_rows) * S_ : nullptr;
  const float* ma   = maskadd + (long)(r / rows_per_batch) * S_;
  float v[2]; float mx = -1e30f;
#pragma unroll
  for (int t = 0; t < 2; ++t) {
    int j = threadIdx.x + t * 256;
    float x = srow[j] * scale;
    if (grow) x *= grow[j];
    x += ma[j];
    v[t] = x; mx = fmaxf(mx, x);
  }
  __shared__ float red[256];
  mx = blk_reduce(mx, red, true);
  float e[2]; float sum = 0.f;
#pragma unroll
  for (int t = 0; t < 2; ++t) { e[t] = __expf(v[t] - mx); sum += e[t]; }
  sum = blk_reduce(sum, red, false);
  float inv = 1.0f / sum;
#pragma unroll
  for (int t = 0; t < 2; ++t)
    probs[r * S_ + threadIdx.x + t * 256] = f2bf(e[t] * inv);
}

// out = LayerNorm(in1 + in2) * w + b ; writes f32 and/or bf16 (H=768 per row).
__global__ __launch_bounds__(256) void k_add_ln(
    const float* __restrict__ in1, long s1,
    const float* __restrict__ in2, long s2,
    const float* __restrict__ w, const float* __restrict__ bb,
    float* __restrict__ o32, long so1,
    bf16* __restrict__ obf, long so2)
{
  long r = blockIdx.x;
  const float* a  = in1 + r * s1;
  const float* b2 = in2 + r * s2;
  float x[3]; float sum = 0.f;
#pragma unroll
  for (int t = 0; t < 3; ++t) {
    int j = threadIdx.x + t * 256;
    x[t] = a[j] + b2[j];
    sum += x[t];
  }
  __shared__ float red[256];
  sum = blk_reduce(sum, red, false);
  float mu = sum * (1.0f / H_);
  float vs = 0.f;
#pragma unroll
  for (int t = 0; t < 3; ++t) { float d = x[t] - mu; vs += d * d; }
  vs = blk_reduce(vs, red, false);
  float rstd = rsqrtf(vs * (1.0f / H_) + 1e-5f);
#pragma unroll
  for (int t = 0; t < 3; ++t) {
    int j = threadIdx.x + t * 256;
    float y = (x[t] - mu) * rstd * w[j] + bb[j];
    if (o32) o32[r * so1 + j] = y;
    if (obf) obf[r * so2 + j] = f2bf(y);
  }
}

// out[b,s,:] = sum_br oo[br,b,s,:] * softmax_br(faw[b,br,s]).  grid B*S.
__global__ __launch_bounds__(256) void k_final(
    const float* __restrict__ oo, const float* __restrict__ faw,
    float* __restrict__ out)
{
  long r = blockIdx.x;
  long b = r / S_, s = r % S_;
  float l[NBR_]; float mx = -1e30f;
#pragma unroll
  for (int br = 0; br < NBR_; ++br) {
    l[br] = faw[(b * NBR_ + br) * MAP_ + s];
    mx = fmaxf(mx, l[br]);
  }
  float sum = 0.f;
#pragma unroll
  for (int br = 0; br < NBR_; ++br) { l[br] = __expf(l[br] - mx); sum += l[br]; }
  float inv = 1.0f / sum;
  for (int j = threadIdx.x; j < H_; j += 256) {
    float acc = 0.f;
#pragma unroll
    for (int br = 0; br < NBR_; ++br)
      acc += oo[((long)br * B_ + b) * S_ * H_ + s * H_ + j] * (l[br] * inv);
    out[r * H_ + j] = acc;
  }
}

// ---------------------------------------------------------------------------
extern "C" void kernel_launch(void* const* d_in, const int* in_sizes, int n_in,
                              void* d_out, int out_size, void* d_ws, size_t ws_size,
                              hipStream_t stream)
{
  (void)in_sizes; (void)n_in; (void)out_size; (void)ws_size;
  const float* x       = (const float*)d_in[0];
  const float* am      = (const float*)d_in[1];
  const float* wt      = (const float*)d_in[2];
  const float* qw      = (const float*)d_in[3];  const float* qb   = (const float*)d_in[4];
  const float* kw      = (const float*)d_in[5];  const float* kb   = (const float*)d_in[6];
  const float* vw      = (const float*)d_in[7];  const float* vb   = (const float*)d_in[8];
  const float* f1w     = (const float*)d_in[9];  const float* f1b  = (const float*)d_in[10];
  const float* f2w     = (const float*)d_in[11]; const float* f2b  = (const float*)d_in[12];
  const float* ln1w    = (const float*)d_in[13]; const float* ln1b = (const float*)d_in[14];
  const float* ln2w    = (const float*)d_in[15]; const float* ln2b = (const float*)d_in[16];
  const float* ca_qw   = (const float*)d_in[17]; const float* ca_qb = (const float*)d_in[18];
  const float* ca_kw   = (const float*)d_in[19]; const float* ca_kb = (const float*)d_in[20];
  const float* ca_vw   = (const float*)d_in[21]; const float* ca_vb = (const float*)d_in[22];
  const float* ca_f1w  = (const float*)d_in[23]; const float* ca_f1b = (const float*)d_in[24];
  const float* ca_f2w  = (const float*)d_in[25]; const float* ca_f2b = (const float*)d_in[26];
  const float* ca_ln1w = (const float*)d_in[27]; const float* ca_ln1b = (const float*)d_in[28];
  const float* ca_ln2w = (const float*)d_in[29]; const float* ca_ln2b = (const float*)d_in[30];
  const float* l1w     = (const float*)d_in[31]; const float* l1b  = (const float*)d_in[32];
  const float* pw      = (const float*)d_in[33]; const float* pb   = (const float*)d_in[34];

  const long HH  = (long)H_ * H_;
  const long HFF = (long)H_ * FF_;
  const long BSH = (long)B_ * S_ * H_;
  const long SS  = (long)S_ * S_;

  char* base = (char*)d_ws;
  size_t off = 0;
  auto alloc = [&](size_t bytes) -> char* {
    char* r = base + off;
    off += (bytes + 255) & ~(size_t)255;
    return r;
  };

  // bf16 transposed weights (persistent within the call; ~150MB, L2-resident)
  bf16* qwT   = (bf16*)alloc((size_t)NBR_ * HH * 2);
  bf16* kwT   = (bf16*)alloc((size_t)NBR_ * HH * 2);
  bf16* vwT   = (bf16*)alloc((size_t)NBR_ * HH * 2);
  bf16* f1wT  = (bf16*)alloc((size_t)NBR_ * HFF * 2);
  bf16* f2wT  = (bf16*)alloc((size_t)NBR_ * HFF * 2);
  bf16* caqT  = (bf16*)alloc((size_t)HH * 2);
  bf16* cakT  = (bf16*)alloc((size_t)HH * 2);
  bf16* cavT  = (bf16*)alloc((size_t)HH * 2);
  bf16* caf1T = (bf16*)alloc((size_t)HH * 2);
  bf16* caf2T = (bf16*)alloc((size_t)HH * 2);
  bf16* l1wT  = (bf16*)alloc((size_t)HH * 2);
  bf16* pwT   = (bf16*)alloc((size_t)MAP_ * H_ * 2);
  // activations & scratch
  bf16*  xb     = (bf16*)alloc((size_t)BSH * 2);
  bf16*  wtb    = (bf16*)alloc((size_t)B_ * H_ * 2);
  bf16*  qcab   = (bf16*)alloc((size_t)B_ * H_ * 2);
  float* gmm    = (float*)alloc((size_t)NBR_ * SS * 4);
  float* madd   = (float*)alloc((size_t)B_ * S_ * 4);
  bf16*  qf     = (bf16*)alloc((size_t)BSH * 2);
  bf16*  kf     = (bf16*)alloc((size_t)BSH * 2);
  bf16*  vf     = (bf16*)alloc((size_t)BSH * 2);
  bf16*  vT     = (bf16*)alloc((size_t)BSH * 2);
  float* scores = (float*)alloc((size_t)B_ * NH_ * SS * 4);
  bf16*  probs  = (bf16*)alloc((size_t)B_ * NH_ * SS * 2);
  float* attno  = (float*)alloc((size_t)BSH * 4);
  float* h1     = (float*)alloc((size_t)BSH * 4);
  bf16*  h1b    = (bf16*)alloc((size_t)BSH * 2);
  bf16*  ffh    = (bf16*)alloc((size_t)B_ * S_ * FF_ * 2);
  float* f2o    = (float*)alloc((size_t)BSH * 4);
  float* oo     = (float*)alloc((size_t)NBR_ * BSH * 4);
  bf16*  oob    = (bf16*)alloc((size_t)NBR_ * BSH * 2);
  bf16*  kca    = (bf16*)alloc((size_t)BSH * 2);
  bf16*  vca    = (bf16*)alloc((size_t)BSH * 2);
  bf16*  vTca   = (bf16*)alloc((size_t)BSH * 2);
  float* ctxca  = (float*)alloc((size_t)B_ * H_ * 4);
  float* qln    = (float*)alloc((size_t)B_ * H_ * 4);
  bf16*  qlnb   = (bf16*)alloc((size_t)B_ * H_ * 2);
  bf16*  ffc    = (bf16*)alloc((size_t)B_ * H_ * 2);
  float* f2c    = (float*)alloc((size_t)B_ * H_ * 4);
  float* wca    = (float*)alloc((size_t)B_ * NBR_ * H_ * 4);
  bf16*  wcab   = (bf16*)alloc((size_t)B_ * NBR_ * H_ * 2);
  bf16*  wfb    = (bf16*)alloc((size_t)B_ * NBR_ * H_ * 2);
  float* fawf   = (float*)alloc((size_t)B_ * NBR_ * MAP_ * 4);

  auto gemm = [&](const bf16* A, long lda, long sA1, long sA2,
                  const bf16* Bt, long ldb, long sB1, long sB2,
                  void* C, long ldc, long sC1, long sC2, bool cbf,
                  const float* bias, bool relu, int M, int N, int K,
                  int nz, int subz) {
    int tiles = ((M + 31) / 32) * (N / 64);
    dim3 g((tiles + 7) / 8, 1, nz);
    k_gemm<<<g, 256, 0, stream>>>(A, lda, sA1, sA2, Bt, ldb, sB1, sB2,
                                  C, ldc, sC1, sC2, cbf ? 1 : 0,
                                  bias, relu ? 1 : 0, M, N, K, subz);
  };
  auto tw = [&](const float* src, bf16* dst, int K, int N, int nz) {
    dim3 g((K + 31) / 32, (N + 31) / 32, nz);
    k_transpose_w<<<g, 256, 0, stream>>>(src, dst, K, N, (long)K * N, (long)K * N);
  };

  // ---- setup: weight transposes, casts, masks -------------------------------
  tw(qw,  qwT,  H_, H_,  NBR_);
  tw(kw,  kwT,  H_, H_,  NBR_);
  tw(vw,  vwT,  H_, H_,  NBR_);
  tw(f1w, f1wT, H_, FF_, NBR_);
  tw(f2w, f2wT, FF_, H_, NBR_);
  tw(ca_qw,  caqT,  H_, H_, 1);
  tw(ca_kw,  cakT,  H_, H_, 1);
  tw(ca_vw,  cavT,  H_, H_, 1);
  tw(ca_f1w, caf1T, H_, H_, 1);
  tw(ca_f2w, caf2T, H_, H_, 1);
  tw(l1w,    l1wT,  H_, H_, 1);
  tw(pw,     pwT,   H_, MAP_, 1);
  k_cvt<<<(BSH + 255) / 256, 256, 0, stream>>>(x, xb, BSH);
  k_bcast_wt<<<(B_ * H_ + 255) / 256, 256, 0, stream>>>(wt, wtb);
  k_gmm<<<dim3(S_, NBR_), 256, 0, stream>>>(gmm);
  k_maskadd<<<(B_ * S_ + 255) / 256, 256, 0, stream>>>(am, madd, B_ * S_);
  // cross-attn query (same for all branches): qca = wt @ ca_qw + ca_qb
  gemm(wtb, H_, 0, 0, caqT, H_, 0, 0, qcab, H_, 0, 0, true,
       ca_qb, false, B_, H_, H_, 1, 1);

  const dim3 gV(S_ / 32, DH_ / 32, B_ * NH_);

  for (int br = 0; br < NBR_; ++br) {
    const bf16* qwTb  = qwT  + (long)br * HH;
    const bf16* kwTb  = kwT  + (long)br * HH;
    const bf16* vwTb  = vwT  + (long)br * HH;
    const bf16* f1wTb = f1wT + (long)br * HFF;
    const bf16* f2wTb = f2wT + (long)br * HFF;

    // ---- self attention ----
    gemm(xb, H_, 0, 0, qwTb, H_, 0, 0, qf, H_, 0, 0, true, qb + (long)br * H_, false, B_ * S_, H_, H_, 1, 1);
    gemm(xb, H_, 0, 0, kwTb, H_, 0, 0, kf, H_, 0, 0, true, kb + (long)br * H_, false, B_ * S_, H_, H_, 1, 1);
    gemm(xb, H_, 0, 0, vwTb, H_, 0, 0, vf, H_, 0, 0, true, vb + (long)br * H_, false, B_ * S_, H_, H_, 1, 1);
    k_transpose_v<<<gV, 256, 0, stream>>>(vf, vT);
    // scores[b,h] = Q Kᵀ : batched over z=(b,h)
    gemm(qf, H_, (long)S_ * H_, DH_, kf, H_, (long)S_ * H_, DH_,
         scores, S_, (long)NH_ * SS, SS, false, nullptr, false,
         S_, S_, DH_, B_ * NH_, NH_);
    k_softmax<<<B_ * NH_ * S_, 256, 0, stream>>>(
        scores, gmm + (long)br * SS, S_, madd, NH_ * S_, 0.125f, probs);
    // ctx = P V : batched, output merged to (B,S,H)
    gemm(probs, S_, (long)NH_ * SS, SS, vT, S_, (long)NH_ * DH_ * S_, (long)DH_ * S_,
         attno, H_, (long)S_ * H_, DH_, false, nullptr, false,
         S_, DH_, S_, B_ * NH_, NH_);
    k_add_ln<<<B_ * S_, 256, 0, stream>>>(x, H_, attno, H_,
        ln1w + (long)br * H_, ln1b + (long)br * H_, h1, H_, h1b, H_);
    // ---- FFN ----
    gemm(h1b, H_, 0, 0, f1wTb, H_, 0, 0, ffh, FF_, 0, 0, true,
         f1b + (long)br * FF_, true, B_ * S_, FF_, H_, 1, 1);
    gemm(ffh, FF_, 0, 0, f2wTb, FF_, 0, 0, f2o, H_, 0, 0, false,
         f2b + (long)br * H_, false, B_ * S_, H_, FF_, 1, 1);
    k_add_ln<<<B_ * S_, 256, 0, stream>>>(h1, H_, f2o, H_,
        ln2w + (long)br * H_, ln2b + (long)br * H_,
        oo + (long)br * BSH, H_, oob + (long)br * BSH, H_);

    // ---- cross attention of weight token vs this branch output ----
    gemm(oob + (long)br * BSH, H_, 0, 0, cakT, H_, 0, 0, kca, H_, 0, 0, true,
         ca_kb, false, B_ * S_, H_, H_, 1, 1);
    gemm(oob + (long)br * BSH, H_, 0, 0, cavT, H_, 0, 0, vca, H_, 0, 0, true,
         ca_vb, false, B_ * S_, H_, H_, 1, 1);
    k_transpose_v<<<gV, 256, 0, stream>>>(vca, vTca);
    gemm(qcab, H_, H_, DH_, kca, H_, (long)S_ * H_, DH_,
         scores, S_, (long)NH_ * S_, S_, false, nullptr, false,
         1, S_, DH_, B_ * NH_, NH_);
    k_softmax<<<B_ * NH_, 256, 0, stream>>>(
        scores, nullptr, 1, madd, NH_, 0.125f, probs);
    gemm(probs, S_, (long)NH_ * S_, S_, vTca, S_, (long)NH_ * DH_ * S_, (long)DH_ * S_,
         ctxca, H_, H_, DH_, false, nullptr, false,
         1, DH_, S_, B_ * NH_, NH_);
    k_add_ln<<<B_, 256, 0, stream>>>(wt, 0, ctxca, H_,
        ca_ln1w, ca_ln1b, qln, H_, qlnb, H_);
    gemm(qlnb, H_, 0, 0, caf1T, H_, 0, 0, ffc, H_, 0, 0, true,
         ca_f1b, true, B_, H_, H_, 1, 1);
    gemm(ffc, H_, 0, 0, caf2T, H_, 0, 0, f2c, H_, 0, 0, false,
         ca_f2b, false, B_, H_, H_, 1, 1);
    k_add_ln<<<B_, 256, 0, stream>>>(qln, H_, f2c, H_,
        ca_ln2w, ca_ln2b,
        wca + (long)br * H_, (long)NBR_ * H_,
        wcab + (long)br * H_, (long)NBR_ * H_);
  }

  // ---- final recombination ----
  gemm(wcab, H_, 0, 0, l1wT, H_, 0, 0, wfb, H_, 0, 0, true,
       l1b, true, B_ * NBR_, H_, H_, 1, 1);
  gemm(wfb, H_, 0, 0, pwT, H_, 0, 0, fawf, MAP_, 0, 0, false,
       pb, false, B_ * NBR_, MAP_, H_, 1, 1);
  k_final<<<B_ * S_, 256, 0, stream>>>(oo, fawf, (float*)d_out);
}
